// TwoReacModel_37074157699116
// MI455X (gfx1250) — matrix operand order repaired and checked by reading, not compiled, not run on hardware
//
#include <hip/hip_runtime.h>
#include <hip/hip_bf16.h>

typedef _Float16 h16;
typedef __attribute__((ext_vector_type(16))) _Float16 v16h;
typedef __attribute__((ext_vector_type(8)))  float    v8f;
typedef __attribute__((ext_vector_type(2)))  float    v2f;

#define WAVES_PER_BLOCK 8
#define T_STEPS 128
#define NBATCH  8192
#define HPAD    68   // padded row (halfs) for h activations in LDS
#define XPAD    17   // padded row (floats) for 16x16 f32 tile in LDS

#define TWOLOG2E 2.8853900817779268f   // 2*log2(e)

#if defined(__has_builtin)
#if __has_builtin(__builtin_amdgcn_wmma_f32_16x16x4_f32)
#define HAVE_WMMA_F32X4 1
#endif
#endif

static __device__ __forceinline__ void wbar() { __builtin_amdgcn_wave_barrier(); }

static __device__ __forceinline__ v8f wmma16(v16h a, v16h b, v8f c) {
  // 8 args: (neg_a, A, neg_b, B, c_mod, C, reuse_a, reuse_b)
  return __builtin_amdgcn_wmma_f32_16x16x32_f16(false, a, false, b, (short)0, c,
                                                false, false);
}

// tanh(c + bias) = 1 - 2/(exp2((c+bias)*2log2e)+1)
// tb = bias*2log2e precomputed -> scale+bias is a single fma feeding v_exp.
static __device__ __forceinline__ float tanh_biased(float c, float tb) {
  float e = __builtin_amdgcn_exp2f(fmaf(c, TWOLOG2E, tb));
  return 1.0f - 2.0f * __builtin_amdgcn_rcpf(e + 1.0f);
}

struct WFrags {
#if HAVE_WMMA_F32X4
  v2f  w1f[4];      // (4x64) f32 B fragments, 4 N-tiles, 2 VGPRs each
#else
  v16h w1[4];       // (4x64) padded to K=32, 4 N-tiles
#endif
  v16h w2[2][4];    // (64x64): 2 K-tiles x 4 N-tiles
  v16h w3[2];       // (64x3) padded to N=16, 2 K-tiles
  float tb1[4];     // b1[n]*2log2e per lane (col of tile j)
  float tb2[4];     // b2[n]*2log2e
  float b30, b31, b32;      // b3 (uniform scalars, added at the end)
  float xm0, xm1, xm2;      // xmean
  float xs0, xs1, xs2;      // xstd
  float g0c, g1c, g2c;      // invtau / xstd[c]   (hoisted reciprocals)
  float um, us;
};

struct F3 { float a, b, c; };

// One evaluation of f(x,u) = fgreybox + fnn for 16 batch rows held by one wave.
// Per-lane state (z0,z1,z2)=x, z3=u for row (lane&15), replicated over halves.
static __device__ __forceinline__ F3 eval_f(
    float z0, float z1, float z2, float z3,
    const WFrags& W, h16 (*hb)[HPAD], float (*xb)[XPAD],
    int half, int col)
{
  const int row = col;
  const bool lo = (half == 0);
  const v8f czero = {};   // lowered as inline-0 SRC2 on the first WMMA of a chain

  // ---------- layer 1 ----------
  v8f c1[4];
#if HAVE_WMMA_F32X4
  // A 16x4 f32: lanes 0-15 hold K=0,1 in v0,v1; lanes 16-31 hold K=2,3.
  v2f a1;
  a1[0] = lo ? z0 : z2;
  a1[1] = lo ? z1 : z3;
#pragma unroll
  for (int j = 0; j < 4; ++j) {
    c1[j] = __builtin_amdgcn_wmma_f32_16x16x4_f32(false, a1, false, W.w1f[j],
                                                  (short)0, czero, false, false);
#pragma unroll
    for (int v = 0; v < 8; ++v) c1[j][v] = tanh_biased(c1[j][v], W.tb1[j]);
  }
#else
  v16h a1;
#pragma unroll
  for (int i = 0; i < 16; ++i) a1[i] = (h16)0;
  a1[0] = lo ? (h16)z0 : (h16)0;
  a1[1] = lo ? (h16)z1 : (h16)0;
  a1[2] = lo ? (h16)z2 : (h16)0;
  a1[3] = lo ? (h16)z3 : (h16)0;
#pragma unroll
  for (int j = 0; j < 4; ++j) {
    c1[j] = wmma16(a1, W.w1[j], czero);
#pragma unroll
    for (int v = 0; v < 8; ++v) c1[j][v] = tanh_biased(c1[j][v], W.tb1[j]);
  }
#endif

  // store h1 (16x64) to LDS as f16, C layout -> row major
#pragma unroll
  for (int j = 0; j < 4; ++j)
#pragma unroll
    for (int v = 0; v < 8; ++v)
      hb[v + 8 * half][16 * j + col] = (h16)c1[j][v];
  wbar();

  // ---------- layer 2: reload h1 as A fragments ----------
  v16h a2[2];
#pragma unroll
  for (int kt = 0; kt < 2; ++kt)
#pragma unroll
    for (int v = 0; v < 8; ++v) {
      int k = kt * 32 + half * 16 + v * 2;
      a2[kt][2 * v]     = hb[row][k];
      a2[kt][2 * v + 1] = hb[row][k + 1];
    }
  wbar();

  v8f c2[4];
#pragma unroll
  for (int j = 0; j < 4; ++j) {
    c2[j] = wmma16(a2[0], W.w2[0][j], czero);
    c2[j] = wmma16(a2[1], W.w2[1][j], c2[j]);
#pragma unroll
    for (int v = 0; v < 8; ++v) c2[j][v] = tanh_biased(c2[j][v], W.tb2[j]);
  }

  // store h2 to LDS (overwrites h1)
#pragma unroll
  for (int j = 0; j < 4; ++j)
#pragma unroll
    for (int v = 0; v < 8; ++v)
      hb[v + 8 * half][16 * j + col] = (h16)c2[j][v];
  wbar();

  // ---------- layer 3: reload h2 as A fragments ----------
  v16h a3[2];
#pragma unroll
  for (int kt = 0; kt < 2; ++kt)
#pragma unroll
    for (int v = 0; v < 8; ++v) {
      int k = kt * 32 + half * 16 + v * 2;
      a3[kt][2 * v]     = hb[row][k];
      a3[kt][2 * v + 1] = hb[row][k + 1];
    }
  wbar();

  v8f c3;
  c3 = wmma16(a3[0], W.w3[0], czero);
  c3 = wmma16(a3[1], W.w3[1], c3);

  // scatter D tile to LDS f32 so each lane can gather its own row's 3 outputs
#pragma unroll
  for (int v = 0; v < 8; ++v) xb[v + 8 * half][col] = c3[v];
  wbar();
  float f0 = xb[row][0] + W.b30;
  float f1 = xb[row][1] + W.b31;
  float f2 = xb[row][2] + W.b32;
  wbar();

  // ---------- grey-box term (per-lane scalar f32, no divisions) ----------
  float xp0 = fmaf(z0, W.xs0, W.xm0);
  float xp1 = fmaf(z1, W.xs1, W.xm1);
  float xp2 = fmaf(z2, W.xs2, W.xm2);
  float up  = fmaf(z3, W.us,  W.um);

  F3 k;
  k.a = fmaf(up - xp0, W.g0c, f0);   // (up-xp0)/TAU/xs0 + fnn0 + b3
  k.b = fmaf(-xp1,     W.g1c, f1);
  k.c = fmaf(-xp2,     W.g2c, f2);
  return k;
}

__global__ __launch_bounds__(256) void tworeac_rk4_kernel(
    const float* __restrict__ u,   const float* __restrict__ x0,
    const float* __restrict__ W1,  const float* __restrict__ b1,
    const float* __restrict__ W2,  const float* __restrict__ b2,
    const float* __restrict__ W3,  const float* __restrict__ b3,
    const float* __restrict__ xmean, const float* __restrict__ xstd,
    const float* __restrict__ umean, const float* __restrict__ ustd,
    float* __restrict__ out)
{
  __shared__ h16   hbuf[WAVES_PER_BLOCK][16][HPAD];
  __shared__ float xbuf[WAVES_PER_BLOCK][16][XPAD];

  const int lane = threadIdx.x & 31;
  const int wv   = threadIdx.x >> 5;
  const int half = lane >> 4;
  const int col  = lane & 15;
  const int row  = col;
  const int rowBase = (blockIdx.x * WAVES_PER_BLOCK + wv) << 4;

  h16   (*hb)[HPAD] = hbuf[wv];
  float (*xb)[XPAD] = xbuf[wv];

  // -------- build loop-invariant weight fragments in VGPRs --------
  WFrags W;
#pragma unroll
  for (int j = 0; j < 4; ++j) {
    const int n = 16 * j + col;
#if HAVE_WMMA_F32X4
    // B 4x16 f32: lane=col; lanes 0-15 hold K=0,1; lanes 16-31 hold K=2,3.
#pragma unroll
    for (int v = 0; v < 2; ++v) {
      int k = 2 * half + v;
      W.w1f[j][v] = W1[k * 64 + n];
    }
#else
#pragma unroll
    for (int v = 0; v < 8; ++v) {
      int k = 16 * half + 2 * v;
      W.w1[j][2 * v]     = (k     < 4) ? (h16)W1[k * 64 + n]       : (h16)0;
      W.w1[j][2 * v + 1] = (k + 1 < 4) ? (h16)W1[(k + 1) * 64 + n] : (h16)0;
    }
#endif
#pragma unroll
    for (int kt = 0; kt < 2; ++kt)
#pragma unroll
      for (int v = 0; v < 8; ++v) {
        int k = 32 * kt + 16 * half + 2 * v;
        W.w2[kt][j][2 * v]     = (h16)W2[k * 64 + n];
        W.w2[kt][j][2 * v + 1] = (h16)W2[(k + 1) * 64 + n];
      }
    W.tb1[j] = b1[n] * TWOLOG2E;
    W.tb2[j] = b2[n] * TWOLOG2E;
  }
#pragma unroll
  for (int kt = 0; kt < 2; ++kt)
#pragma unroll
    for (int v = 0; v < 8; ++v) {
      int k = 32 * kt + 16 * half + 2 * v;
      W.w3[kt][2 * v]     = (col < 3) ? (h16)W3[k * 3 + col]       : (h16)0;
      W.w3[kt][2 * v + 1] = (col < 3) ? (h16)W3[(k + 1) * 3 + col] : (h16)0;
    }
  W.b30 = b3[0]; W.b31 = b3[1]; W.b32 = b3[2];
  W.xm0 = xmean[0]; W.xm1 = xmean[1]; W.xm2 = xmean[2];
  W.xs0 = xstd[0];  W.xs1 = xstd[1];  W.xs2 = xstd[2];
  const float invtau = 1.0f / 5.0f;   // TAU = 5
  W.g0c = invtau / W.xs0;             // one-time true divides (outside loop)
  W.g1c = invtau / W.xs1;
  W.g2c = invtau / W.xs2;
  W.um  = umean[0]; W.us = ustd[0];

  // -------- load initial state (replicated over the two half-waves) --------
  const size_t gr = (size_t)(rowBase + row);
  float s0 = x0[gr * 3 + 0];
  float s1 = x0[gr * 3 + 1];
  float s2 = x0[gr * 3 + 2];

  // -------- sequential RK4 scan over T --------
  for (int t = 0; t < T_STEPS; ++t) {
    // ys[t] is the PRE-update state
    if (half == 0) {
      size_t ob = (gr * T_STEPS + (size_t)t) * 3;
      out[ob + 0] = s0; out[ob + 1] = s1; out[ob + 2] = s2;
    }
    float uu = u[gr * T_STEPS + (size_t)t];

    F3 k1 = eval_f(s0, s1, s2, uu, W, hb, xb, half, col);
    F3 k2 = eval_f(s0 + 0.5f * k1.a, s1 + 0.5f * k1.b, s2 + 0.5f * k1.c, uu,
                   W, hb, xb, half, col);
    F3 k3 = eval_f(s0 + 0.5f * k2.a, s1 + 0.5f * k2.b, s2 + 0.5f * k2.c, uu,
                   W, hb, xb, half, col);
    F3 k4 = eval_f(s0 + k3.a, s1 + k3.b, s2 + k3.c, uu,
                   W, hb, xb, half, col);

    const float sixth = 1.0f / 6.0f;  // DELTA = 1
    s0 += sixth * (k1.a + 2.0f * k2.a + 2.0f * k3.a + k4.a);
    s1 += sixth * (k1.b + 2.0f * k2.b + 2.0f * k3.b + k4.b);
    s2 += sixth * (k1.c + 2.0f * k2.c + 2.0f * k3.c + k4.c);
  }
}

extern "C" void kernel_launch(void* const* d_in, const int* in_sizes, int n_in,
                              void* d_out, int out_size, void* d_ws, size_t ws_size,
                              hipStream_t stream) {
  const float* u     = (const float*)d_in[0];
  const float* x0    = (const float*)d_in[1];
  const float* W1    = (const float*)d_in[2];
  const float* b1    = (const float*)d_in[3];
  const float* W2    = (const float*)d_in[4];
  const float* b2    = (const float*)d_in[5];
  const float* W3    = (const float*)d_in[6];
  const float* b3    = (const float*)d_in[7];
  const float* xmean = (const float*)d_in[8];
  const float* xstd  = (const float*)d_in[9];
  const float* umean = (const float*)d_in[10];
  const float* ustd  = (const float*)d_in[11];
  float* out = (float*)d_out;

  dim3 grid(NBATCH / (WAVES_PER_BLOCK * 16));  // 64 blocks
  dim3 block(WAVES_PER_BLOCK * 32);            // 256 threads = 8 waves
  tworeac_rk4_kernel<<<grid, block, 0, stream>>>(
      u, x0, W1, b1, W2, b2, W3, b3, xmean, xstd, umean, ustd, out);
}
